// LinearMultiHeadSelfAttention1D_5471788335638
// MI455X (gfx1250) — matrix operand order repaired and checked by reading, not compiled, not run on hardware
//
#include <hip/hip_runtime.h>

// MI455X / gfx1250 linear multi-head attention, bf16 WMMA pipeline.
// All matrix math on v_wmma_f32_16x16x32_bf16 (f32 accumulate).
// - Fragments: contiguous 16B/32B per-lane runs (ds_load_b128/global_load_b128)
// - Staging: pair-packed v_cvt_pk_bf16_f32 + ds_store_b32
// - K3 computes num^T = S^T @ q^T so den is per-lane and attn stores are packed

typedef __attribute__((ext_vector_type(16))) __bf16 v16bf;
typedef __attribute__((ext_vector_type(8)))  __bf16 v8bf;
typedef __attribute__((ext_vector_type(2)))  __bf16 v2bf;
typedef __attribute__((ext_vector_type(8)))  float  v8f;

#define B_   4
#define C_   512
#define N_   16384
#define H_   8
#define D_   64
#define C3_  1536
#define EPS_ 1e-6f
#define CHUNK_ 1024

__device__ __forceinline__ float phi_f(float x) {
  // elu(x)+1 == x+1 (x>0) else exp(x)
  return x > 0.f ? x + 1.f : __expf(x);
}

__device__ __forceinline__ v16bf cat8(v8bf lo, v8bf hi) {
  return __builtin_shufflevector(lo, hi, 0, 1, 2, 3, 4, 5, 6, 7,
                                 8, 9, 10, 11, 12, 13, 14, 15);
}

__device__ __forceinline__ v8f wmma_bf16(v16bf a, v16bf b, v8f c) {
  return __builtin_amdgcn_wmma_f32_16x16x32_bf16(false, a, false, b,
                                                 (short)0, c, false, false);
}

// ---------------------------------------------------------------------------
// Kernel 1: qkv = x^T @ W_qkv + b ; phi on q,k.
//   q  -> [B,H,N,D] bf16   (row-major, feeds K3 B-fragments)
//   kT -> [B,H,D,N] bf16   (transposed: packed v8bf stores here; contiguous
//   vT -> [B,H,D,N] bf16    global fragment runs in K2)
// Block 256 thr (8 waves), tile 128n x 64j, K-step 32 over C=512.
// LDS tiles K-minor: As[nn][kk], Bs[jj][kk], pitch 48 halves (96B, 32B-mult).
// Staging pair-packs adjacent K rows -> v_cvt_pk_bf16_f32 + ds_store_b32.
// ---------------------------------------------------------------------------
__global__ __launch_bounds__(256)
void k_qkv_phi(const float* __restrict__ x, const float* __restrict__ Wqkv,
               const float* __restrict__ bqkv,
               __bf16* __restrict__ qp, __bf16* __restrict__ kT,
               __bf16* __restrict__ vT) {
  __shared__ __attribute__((aligned(32))) __bf16 As[128 * 48];
  __shared__ __attribute__((aligned(32))) __bf16 Bs[64 * 48];
  const int b  = blockIdx.y;
  const int nTiles = N_ / 128;
  const int n0 = (blockIdx.x % nTiles) * 128;
  const int j0 = (blockIdx.x / nTiles) * 64;
  const int tid = threadIdx.x, lane = tid & 31, w = tid >> 5;
  const int hi = lane >> 4, lo = lane & 15;
  const int wn = w >> 1, wj = w & 1;

  v8f acc[2][2] = {};

  for (int k0 = 0; k0 < C_; k0 += 32) {
    // stage x tile -> As[nn][kk]; thread owns 4 n, one even-k pair per l
    {
      const int nn4 = (tid & 31) * 4;
      const int kp  = tid >> 5;  // 0..7
      #pragma unroll
      for (int l = 0; l < 2; ++l) {
        const int kk = (l * 8 + kp) * 2;
        const float* xb = &x[(size_t)b * C_ * N_ +
                             (size_t)(k0 + kk) * N_ + n0 + nn4];
        float4 r0 = *(const float4*)xb;
        float4 r1 = *(const float4*)(xb + N_);
        const float a0[4] = {r0.x, r0.y, r0.z, r0.w};
        const float a1[4] = {r1.x, r1.y, r1.z, r1.w};
        #pragma unroll
        for (int q = 0; q < 4; ++q) {
          v2bf p; p[0] = (__bf16)a0[q]; p[1] = (__bf16)a1[q];
          *(v2bf*)&As[(nn4 + q) * 48 + kk] = p;
        }
      }
    }
    // stage W tile -> Bs[jj][kk]; thread owns 4 j, one even-k pair
    {
      const int jj4 = (tid & 15) * 4;
      const int kk  = (tid >> 4) * 2;  // 0..30
      const float* wb = &Wqkv[(size_t)(k0 + kk) * C3_ + j0 + jj4];
      float4 r0 = *(const float4*)wb;
      float4 r1 = *(const float4*)(wb + C3_);
      const float a0[4] = {r0.x, r0.y, r0.z, r0.w};
      const float a1[4] = {r1.x, r1.y, r1.z, r1.w};
      #pragma unroll
      for (int q = 0; q < 4; ++q) {
        v2bf p; p[0] = (__bf16)a0[q]; p[1] = (__bf16)a1[q];
        *(v2bf*)&Bs[(jj4 + q) * 48 + kk] = p;
      }
    }
    if (k0 + 32 < C_)  // pull next x tile toward L2/L0 (global_prefetch_b8)
      __builtin_prefetch(&x[(size_t)b * C_ * N_ +
                            (size_t)(k0 + 32 + (tid >> 5)) * N_ + n0 +
                            (tid & 31) * 4], 0, 1);
    __syncthreads();

    v16bf a[2], bf[2];
    #pragma unroll
    for (int i = 0; i < 2; ++i) {
      const __bf16* row = &As[(wn * 32 + i * 16 + lo) * 48];
      a[i] = *(const v16bf*)(row + hi * 16);
    }
    #pragma unroll
    for (int j = 0; j < 2; ++j)
      bf[j] = *(const v16bf*)&Bs[(wj * 32 + j * 16 + lo) * 48 + hi * 16];

    #pragma unroll
    for (int i = 0; i < 2; ++i)
      #pragma unroll
      for (int j = 0; j < 2; ++j)
        acc[i][j] = wmma_bf16(a[i], bf[j], acc[i][j]);
    __syncthreads();
  }

  // epilogue: bias, phi(q,k). Lane owns one output column jo and 8
  // consecutive n rows -> kT/vT stores are packed 16B v8bf.
  #pragma unroll
  for (int i = 0; i < 2; ++i) {
    #pragma unroll
    for (int j = 0; j < 2; ++j) {
      const int jo = j0 + wj * 32 + j * 16 + lo;
      const float bias = bqkv[jo];
      const int typ = jo >> 9;  // 0=q, 1=k, 2=v
      const int c = jo & 511;
      const int h = c >> 6, d = c & 63;
      const int nb = n0 + wn * 32 + i * 16 + hi * 8;  // 8 consecutive n
      if (typ == 0) {
        #pragma unroll
        for (int r = 0; r < 8; ++r)
          qp[((size_t)(b * H_ + h) * N_ + nb + r) * D_ + d] =
              (__bf16)phi_f(acc[i][j][r] + bias);
      } else if (typ == 1) {
        v8bf pk;
        #pragma unroll
        for (int r = 0; r < 8; ++r) pk[r] = (__bf16)phi_f(acc[i][j][r] + bias);
        *(v8bf*)(kT + ((size_t)(b * H_ + h) * D_ + d) * N_ + nb) = pk;
      } else {
        v8bf pk;
        #pragma unroll
        for (int r = 0; r < 8; ++r) pk[r] = (__bf16)(acc[i][j][r] + bias);
        *(v8bf*)(vT + ((size_t)(b * H_ + h) * D_ + d) * N_ + nb) = pk;
      }
    }
  }
}

// ---------------------------------------------------------------------------
// Kernel 2: S[bh] += k_phi^T @ v over an N-chunk (f32 atomics), k_sum partials.
// No LDS: kT/vT are [B,H,D,N], so A fragments are two contiguous v8bf global
// loads and B fragments one contiguous v16bf (2x b128) global load.
// Wave w owns one dd row-group (dd0=(w>>1)*16) and two e tiles.
// ---------------------------------------------------------------------------
__global__ __launch_bounds__(256)
void k_kv_state(const __bf16* __restrict__ kT, const __bf16* __restrict__ vT,
                float* __restrict__ Sg, float* __restrict__ ksum) {
  const int bh = blockIdx.y;
  const int nbase = blockIdx.x * CHUNK_;
  const int tid = threadIdx.x, lane = tid & 31, w = tid >> 5;
  const int hi = lane >> 4, lo = lane & 15;

  const int dd0 = (w >> 1) * 16;
  const int e0a = ((w & 1) * 2 + 0) * 16;
  const int e0b = ((w & 1) * 2 + 1) * 16;

  const __bf16* arow  = kT + ((size_t)bh * D_ + dd0 + lo) * N_ + nbase;
  const __bf16* brow0 = vT + ((size_t)bh * D_ + e0a + lo) * N_ + nbase;
  const __bf16* brow1 = vT + ((size_t)bh * D_ + e0b + lo) * N_ + nbase;

  v8f acc0 = {}, acc1 = {};
  for (int ni = 0; ni < CHUNK_; ni += 32) {
    v16bf a = *(const v16bf*)(arow + ni + hi * 16);
    v16bf b0 = *(const v16bf*)(brow0 + ni + hi * 16);
    v16bf b1 = *(const v16bf*)(brow1 + ni + hi * 16);
    acc0 = wmma_bf16(a, b0, acc0);
    acc1 = wmma_bf16(a, b1, acc1);
  }
  #pragma unroll
  for (int r = 0; r < 8; ++r) {
    int dd = dd0 + r + hi * 8;
    atomicAdd(&Sg[((size_t)bh * D_ + dd) * D_ + e0a + lo], acc0[r]);
    atomicAdd(&Sg[((size_t)bh * D_ + dd) * D_ + e0b + lo], acc1[r]);
  }
  // k_sum partials: thread -> (d = tid>>2, 256-wide contiguous n segment),
  // read as packed v8bf (16B) chunks.
  {
    int d = tid >> 2, seg = tid & 3;
    const __bf16* kr = kT + ((size_t)bh * D_ + d) * N_ + nbase + seg * 256;
    float s = 0.f;
    #pragma unroll 4
    for (int i = 0; i < 32; ++i) {
      v8bf kv = *(const v8bf*)(kr + i * 8);
      #pragma unroll
      for (int r = 0; r < 8; ++r) s += (float)kv[r];
    }
    atomicAdd(&ksum[bh * D_ + d], s);
  }
}

// ---------------------------------------------------------------------------
// Kernel 3: num^T = S^T @ q^T (WMMA), den per-lane, attn = num/den -> [B,N,C].
// A = S^T from LDS (SsT[e][dd], pitch 80 halves = 160B): two v8bf ds reads.
// B = q^T read straight from row-major q: one contiguous v16bf per lane.
// Output lane owns one n and 8 consecutive e -> packed v8bf attn stores.
// den computed from the already-loaded B fragments + one __shfl_xor(16);
// ksum read at per-lane-half offsets (it*32 + hi*16 + e2).
// ---------------------------------------------------------------------------
__global__ __launch_bounds__(256)
void k_apply(const __bf16* __restrict__ qp, const float* __restrict__ Sg,
             const float* __restrict__ ksum, __bf16* __restrict__ attn) {
  __shared__ __attribute__((aligned(32))) __bf16 SsT[64 * 80];  // [e][dd]
  const int bh = blockIdx.y;
  const int b = bh >> 3, h = bh & 7;
  const int tid = threadIdx.x, lane = tid & 31, w = tid >> 5;
  const int hi = lane >> 4, lo = lane & 15;

  // stage S^T (pair-packed along dd)
  {
    const int e4 = (tid & 15) * 4;
    const int dp = tid >> 4;  // 0..15
    #pragma unroll
    for (int l = 0; l < 2; ++l) {
      const int dd = (l * 16 + dp) * 2;
      const float* sb = &Sg[((size_t)bh * D_ + dd) * D_ + e4];
      float4 r0 = *(const float4*)sb;
      float4 r1 = *(const float4*)(sb + D_);
      const float s0[4] = {r0.x, r0.y, r0.z, r0.w};
      const float s1[4] = {r1.x, r1.y, r1.z, r1.w};
      #pragma unroll
      for (int q = 0; q < 4; ++q) {
        v2bf p; p[0] = (__bf16)s0[q]; p[1] = (__bf16)s1[q];
        *(v2bf*)&SsT[(e4 + q) * 80 + dd] = p;
      }
    }
  }
  __syncthreads();

  const int n = blockIdx.x * 128 + w * 16 + lo;
  const __bf16* qrow = qp + ((size_t)bh * N_ + n) * D_;

  v8f acc[4] = {};
  float denp = 0.f;
  #pragma unroll
  for (int it = 0; it < 2; ++it) {
    v16bf bq = *(const v16bf*)(qrow + it * 32 + hi * 16);  // q^T B-fragment
    // den partial: this lane's 16 d values are it*32 + hi*16 + e2
    #pragma unroll
    for (int e2 = 0; e2 < 16; ++e2)
      denp += (float)bq[e2] * ksum[bh * D_ + it * 32 + hi * 16 + e2];
    #pragma unroll
    for (int tj = 0; tj < 4; ++tj) {
      const __bf16* srow = &SsT[(tj * 16 + lo) * 80 + it * 32];
      v16bf a = *(const v16bf*)(srow + hi * 16);
      acc[tj] = wmma_bf16(a, bq, acc[tj]);
    }
  }
  const float den = denp + __shfl_xor(denp, 16, 32) + EPS_;
  const float rden = 1.f / den;

  #pragma unroll
  for (int tj = 0; tj < 4; ++tj) {
    v8bf pk;
    #pragma unroll
    for (int r = 0; r < 8; ++r) pk[r] = (__bf16)(acc[tj][r] * rden);
    *(v8bf*)&attn[((size_t)b * N_ + n) * C_ + h * 64 + tj * 16 + hi * 8] = pk;
  }
}

// ---------------------------------------------------------------------------
// Kernel 4: out[b][cout][n] = attn[b][n][:] @ W_proj[:,cout] + b_proj[cout].
// A fragments contiguous from row-major attn; W staged K-minor (pair-packed).
// Output [B,C,N]: each lane owns one cout and 8 consecutive n -> two packed
// float4 (b128) stores.
// ---------------------------------------------------------------------------
__global__ __launch_bounds__(256)
void k_proj(const __bf16* __restrict__ attn, const float* __restrict__ Wp,
            const float* __restrict__ bp, float* __restrict__ out) {
  __shared__ __attribute__((aligned(32))) __bf16 BsT[64 * 48];  // [jj][kk]
  const int b = blockIdx.y;
  const int jBlocks = C_ / 64;
  const int n0 = (blockIdx.x / jBlocks) * 128;
  const int j0 = (blockIdx.x % jBlocks) * 64;
  const int tid = threadIdx.x, lane = tid & 31, w = tid >> 5;
  const int hi = lane >> 4, lo = lane & 15;
  const int wn = w >> 1, wj = w & 1;

  v8f acc[2][2] = {};
  for (int k0 = 0; k0 < C_; k0 += 32) {
    {
      const int jj4 = (tid & 15) * 4;
      const int kk  = (tid >> 4) * 2;
      const float* wb = &Wp[(size_t)(k0 + kk) * C_ + j0 + jj4];
      float4 r0 = *(const float4*)wb;
      float4 r1 = *(const float4*)(wb + C_);
      const float a0[4] = {r0.x, r0.y, r0.z, r0.w};
      const float a1[4] = {r1.x, r1.y, r1.z, r1.w};
      #pragma unroll
      for (int q = 0; q < 4; ++q) {
        v2bf p; p[0] = (__bf16)a0[q]; p[1] = (__bf16)a1[q];
        *(v2bf*)&BsT[(jj4 + q) * 48 + kk] = p;
      }
    }
    __syncthreads();
    v16bf a[2];
    #pragma unroll
    for (int i = 0; i < 2; ++i) {
      const __bf16* arow =
          attn + ((size_t)b * N_ + n0 + wn * 32 + i * 16 + lo) * C_ + k0;
      a[i] = *(const v16bf*)(arow + hi * 16);
    }
    #pragma unroll
    for (int i = 0; i < 2; ++i)
      #pragma unroll
      for (int j = 0; j < 2; ++j) {
        v16bf bf = *(const v16bf*)&BsT[(wj * 32 + j * 16 + lo) * 48 + hi * 16];
        acc[i][j] = wmma_bf16(a[i], bf, acc[i][j]);
      }
    __syncthreads();
  }
  #pragma unroll
  for (int i = 0; i < 2; ++i)
    #pragma unroll
    for (int j = 0; j < 2; ++j) {
      const int cout = j0 + wj * 32 + j * 16 + lo;
      const float bias = bp[cout];
      const int nb = n0 + wn * 32 + i * 16 + hi * 8;  // 8 consecutive n
      float* op = out + (size_t)b * C_ * N_ + (size_t)cout * N_ + nb;
      float4 s0 = make_float4(acc[i][j][0] + bias, acc[i][j][1] + bias,
                              acc[i][j][2] + bias, acc[i][j][3] + bias);
      float4 s1 = make_float4(acc[i][j][4] + bias, acc[i][j][5] + bias,
                              acc[i][j][6] + bias, acc[i][j][7] + bias);
      *(float4*)(op + 0) = s0;
      *(float4*)(op + 4) = s1;
    }
}

extern "C" void kernel_launch(void* const* d_in, const int* in_sizes, int n_in,
                              void* d_out, int out_size, void* d_ws, size_t ws_size,
                              hipStream_t stream) {
  (void)in_sizes; (void)n_in; (void)out_size; (void)ws_size;
  const float* x    = (const float*)d_in[0];
  const float* Wqkv = (const float*)d_in[1];
  const float* bqkv = (const float*)d_in[2];
  const float* Wp   = (const float*)d_in[3];
  const float* bp   = (const float*)d_in[4];
  float* out = (float*)d_out;

  // workspace layout (bf16 intermediates + f32 state), ~269 MB total
  const size_t tEl = (size_t)B_ * N_ * C_;  // 33.55M elems per tensor
  __bf16* qp   = (__bf16*)d_ws;             // [B,H,N,D]
  __bf16* kT   = qp + tEl;                  // [B,H,D,N]
  __bf16* vT   = kT + tEl;                  // [B,H,D,N]
  __bf16* attn = vT + tEl;                  // [B,N,C]
  float*  Sg   = (float*)(attn + tEl);      // [B*H,64,64]
  float*  ksum = Sg + (size_t)B_ * H_ * D_ * D_;  // [B*H,64]

  hipMemsetAsync(Sg, 0,
                 ((size_t)B_ * H_ * D_ * D_ + (size_t)B_ * H_ * D_) * sizeof(float),
                 stream);

  dim3 blk(256);
  k_qkv_phi<<<dim3((N_ / 128) * (C3_ / 64), B_), blk, 0, stream>>>(x, Wqkv, bqkv,
                                                                   qp, kT, vT);
  k_kv_state<<<dim3(N_ / CHUNK_, B_ * H_), blk, 0, stream>>>(kT, vT, Sg, ksum);
  k_apply<<<dim3(N_ / 128, B_ * H_), blk, 0, stream>>>(qp, Sg, ksum, attn);
  k_proj<<<dim3((N_ / 128) * (C_ / 64), B_), blk, 0, stream>>>(attn, Wp, bp, out);
}
